// LSTM_ODE_48842368090279
// MI455X (gfx1250) — matrix operand order repaired and checked by reading, not compiled, not run on hardware
//
#include <hip/hip_runtime.h>
#include <hip/hip_bf16.h>
#include <math.h>

// ---------------------------------------------------------------------------
// LSTM -> Neural ODE (dopri5, fixed 40 steps) -> linear head, for gfx1250.
// All GEMMs via v_wmma_f32_16x16x32_bf16 (bf16 inputs, f32 accumulate).
// Weights are pre-converted once into bf16 [N,K] row-major, read as the WMMA
// B operand in contiguous k order, same as A (lane n holds 16 contiguous K halfs).
// Wave tile 32x64 -> 8 WMMAs per 12 b128 loads in the inner loop.
// ---------------------------------------------------------------------------

typedef __attribute__((ext_vector_type(16))) __bf16 v16bf;
typedef __attribute__((ext_vector_type(8)))  __bf16 v8bf;
typedef __attribute__((ext_vector_type(8)))  float  v8f;

#define B_   256
#define L_   512
#define I_   64
#define H_   512
#define F_   24
#define G4H  2048          // 4*H
#define KC   576           // I + H (concat GEMM K)
#define BH   131072        // B_*H_
#define DT_  (1.0f/40.0f)

__device__ __forceinline__ float sigf(float v) { return 1.0f / (1.0f + __expf(-v)); }

// ---------------------------------------------------------------------------
// Generic bf16 WMMA GEMM:  C[M,N] = epi( A[M,K] * W[N,K]^T + bias[N] )
//   A row-major bf16 (lda halfs), W row-major [N,K] bf16 (ldw halfs).
//   Block = WM*WN waves; wave tile = (TM*16) x (TN*16).
//   EPI==0: store f32 to Cf.  EPI==1: store bf16(tanh(.)) to Cb.
// ---------------------------------------------------------------------------
template<int WM, int WN, int TM, int TN, int EPI>
__global__ void gemm_bf16_kernel(const __hip_bfloat16* __restrict__ A, int lda,
                                 const __hip_bfloat16* __restrict__ W, int ldw,
                                 const float* __restrict__ bias,
                                 float* __restrict__ Cf,
                                 __hip_bfloat16* __restrict__ Cb,
                                 int ldc, int K, int nvalid)
{
    const int tid  = threadIdx.x;
    const int lane = tid & 31;
    const int wave = tid >> 5;
    const int wm   = wave / WN;
    const int wn   = wave % WN;
    const int m0   = blockIdx.y * (WM * TM * 16) + wm * (TM * 16);
    const int n0   = blockIdx.x * (WN * TN * 16) + wn * (TN * 16);

    // A-fragment addressing (16-bit A 16x32): lanes 0-15 hold K 0-15,
    // lanes 16-31 hold K 16-31 -> one contiguous 16-half chunk (matches B).
    const int arow  = lane & 15;
    const int aksel = (lane >> 4) * 16;
    // B-fragment addressing (16-bit B 32x16): lane n%16 holds 16 contiguous K
    // values; lanes 16-31 cover K 16..31 -> one contiguous 16-half chunk.
    const int bcol  = lane & 15;
    const int bksel = (lane >> 4) * 16;

    const __bf16* Ab = (const __bf16*)A;
    const __bf16* Wb = (const __bf16*)W;

    v8f acc[TM][TN] = {};

    for (int k0 = 0; k0 < K; k0 += 32) {
        v16bf afrag[TM], bfrag[TN];
#pragma unroll
        for (int tm = 0; tm < TM; ++tm) {
            const __bf16* pa = Ab + (size_t)(m0 + tm * 16 + arow) * lda + k0 + aksel;
            v8bf lo = *(const v8bf*)(pa);
            v8bf hi = *(const v8bf*)(pa + 8);
            afrag[tm] = __builtin_shufflevector(lo, hi,
                0, 1, 2, 3, 4, 5, 6, 7, 8, 9, 10, 11, 12, 13, 14, 15);
        }
#pragma unroll
        for (int tn = 0; tn < TN; ++tn) {
            const __bf16* pb = Wb + (size_t)(n0 + tn * 16 + bcol) * ldw + k0 + bksel;
            bfrag[tn] = *(const v16bf*)(pb);
            // Speculative prefetch of the next K tile (silently dropped if
            // past the mapping); unconditional keeps the loop branch-free.
            __builtin_prefetch((const void*)(pb + 32), 0, 3);
        }
#pragma unroll
        for (int tm = 0; tm < TM; ++tm)
#pragma unroll
            for (int tn = 0; tn < TN; ++tn)
                acc[tm][tn] = __builtin_amdgcn_wmma_f32_16x16x32_bf16(
                    false, afrag[tm], false, bfrag[tn],
                    (short)0, acc[tm][tn], false, false);
    }

    // C/D layout: VGPR r, lanes 0-15 -> row r, lanes 16-31 -> row r+8; col = lane%16.
#pragma unroll
    for (int tm = 0; tm < TM; ++tm) {
        const int mrow = m0 + tm * 16 + ((lane >> 4) * 8);
#pragma unroll
        for (int tn = 0; tn < TN; ++tn) {
            const int n = n0 + tn * 16 + (lane & 15);
            if (n < nvalid) {
                const float bv = bias[n];
#pragma unroll
                for (int r = 0; r < 8; ++r) {
                    float v = acc[tm][tn][r] + bv;
                    if (EPI == 0) Cf[(size_t)(mrow + r) * ldc + n] = v;
                    else          Cb[(size_t)(mrow + r) * ldc + n] =
                                      __float2bfloat16(tanhf(v));
                }
            }
        }
    }
}

// --------------------------- weight conversion -----------------------------
__global__ void convert_wcat_kernel(const float* __restrict__ W_ih,
                                    const float* __restrict__ W_hh,
                                    const float* __restrict__ b_ih,
                                    const float* __restrict__ b_hh,
                                    __hip_bfloat16* __restrict__ Wcat,
                                    float* __restrict__ biascat)
{
    int idx = blockIdx.x * blockDim.x + threadIdx.x;
    if (idx >= G4H * KC) return;
    int n = idx / KC, j = idx % KC;
    float v = (j < I_) ? W_ih[n * I_ + j] : W_hh[n * H_ + (j - I_)];
    Wcat[idx] = __float2bfloat16(v);
    if (j == 0) biascat[n] = b_ih[n] + b_hh[n];
}

__global__ void convert_f32_bf16_kernel(const float* __restrict__ src,
                                        __hip_bfloat16* __restrict__ dst, int n)
{
    int idx = blockIdx.x * blockDim.x + threadIdx.x;
    if (idx < n) dst[idx] = __float2bfloat16(src[idx]);
}

__global__ void convert_wout_kernel(const float* __restrict__ Wout,
                                    __hip_bfloat16* __restrict__ dst)
{
    int idx = blockIdx.x * blockDim.x + threadIdx.x;   // 32*512
    if (idx >= 32 * H_) return;
    int n = idx / H_, k = idx % H_;
    dst[idx] = __float2bfloat16(n < F_ ? Wout[n * H_ + k] : 0.0f);
}

// ----------------------------- LSTM kernels --------------------------------
// Abuf (bf16 [B, 576]) = [ x_t | h ] — the A matrix of the fused gates GEMM.
__global__ void lstm_init_kernel(const float* __restrict__ x,
                                 __hip_bfloat16* __restrict__ Abuf,
                                 float* __restrict__ c)
{
    int idx = blockIdx.x * blockDim.x + threadIdx.x;
    if (idx >= B_ * KC) return;
    int b = idx / KC, j = idx % KC;
    if (j < I_) {
        Abuf[idx] = __float2bfloat16(x[(size_t)b * L_ * I_ + j]);   // x[b,0,j]
    } else {
        Abuf[idx] = __float2bfloat16(0.0f);                         // h0 = 0
        c[b * H_ + (j - I_)] = 0.0f;                                // c0 = 0
    }
}

__global__ void lstm_update_kernel(const float* __restrict__ gates,
                                   float* __restrict__ c,
                                   __hip_bfloat16* __restrict__ Abuf,
                                   float* __restrict__ hout,
                                   const float* __restrict__ x, int t)
{
    int idx = blockIdx.x * blockDim.x + threadIdx.x;   // B_*H_
    if (idx >= BH) return;
    int b = idx >> 9, u = idx & (H_ - 1);
    const float* g = gates + (size_t)b * G4H;
    float gi = g[u], gf = g[H_ + u], gg = g[2 * H_ + u], go = g[3 * H_ + u];
    float cn = sigf(gf) * c[idx] + sigf(gi) * tanhf(gg);
    c[idx] = cn;
    float h = sigf(go) * tanhf(cn);
    Abuf[b * KC + I_ + u] = __float2bfloat16(h);
    hout[idx] = h;                                     // last write = h_L
    if (u < I_ && (t + 1) < L_)                        // stage x_{t+1}
        Abuf[b * KC + u] =
            __float2bfloat16(x[(size_t)b * L_ * I_ + (size_t)(t + 1) * I_ + u]);
}

// ------------------------------ ODE kernels --------------------------------
// As = bf16( y + DT * sum_j coef_j * k_j ), cnt in [0,5]
__global__ void ode_combine_kernel(const float* __restrict__ y,
                                   const float* __restrict__ kb,
                                   __hip_bfloat16* __restrict__ As,
                                   float c1, float c2, float c3, float c4,
                                   float c5, int cnt)
{
    int idx = blockIdx.x * blockDim.x + threadIdx.x;
    if (idx >= BH) return;
    float acc = y[idx];
    if (cnt > 0) acc += DT_ * c1 * kb[0 * BH + idx];
    if (cnt > 1) acc += DT_ * c2 * kb[1 * BH + idx];
    if (cnt > 2) acc += DT_ * c3 * kb[2 * BH + idx];
    if (cnt > 3) acc += DT_ * c4 * kb[3 * BH + idx];
    if (cnt > 4) acc += DT_ * c5 * kb[4 * BH + idx];
    As[idx] = __float2bfloat16(acc);
}

__global__ void ode_final_kernel(float* __restrict__ y, const float* __restrict__ kb)
{
    int idx = blockIdx.x * blockDim.x + threadIdx.x;
    if (idx >= BH) return;
    const float B1 = 35.0f / 384.0f, B3 = 500.0f / 1113.0f, B4 = 125.0f / 192.0f,
                B5 = -2187.0f / 6784.0f, B6 = 11.0f / 84.0f;
    y[idx] += DT_ * (B1 * kb[0 * BH + idx] + B3 * kb[2 * BH + idx] +
                     B4 * kb[3 * BH + idx] + B5 * kb[4 * BH + idx] +
                     B6 * kb[5 * BH + idx]);
}

// ---------------------------------------------------------------------------
extern "C" void kernel_launch(void* const* d_in, const int* in_sizes, int n_in,
                              void* d_out, int out_size, void* d_ws, size_t ws_size,
                              hipStream_t stream)
{
    (void)in_sizes; (void)n_in; (void)out_size; (void)ws_size;

    const float* x    = (const float*)d_in[0];
    const float* W_ih = (const float*)d_in[1];
    const float* W_hh = (const float*)d_in[2];
    const float* b_ih = (const float*)d_in[3];
    const float* b_hh = (const float*)d_in[4];
    const float* W1   = (const float*)d_in[5];
    const float* b1   = (const float*)d_in[6];
    const float* W2   = (const float*)d_in[7];
    const float* b2   = (const float*)d_in[8];
    const float* Wout = (const float*)d_in[9];
    const float* bout = (const float*)d_in[10];

    char* base = (char*)d_ws;
    size_t off = 0;
    auto take = [&](size_t bytes) -> char* {
        char* r = base + off;
        off += (bytes + 255) & ~(size_t)255;
        return r;
    };

    __hip_bfloat16* Wcat    = (__hip_bfloat16*)take((size_t)G4H * KC * 2);
    float*          biascat = (float*)         take((size_t)G4H * 4);
    __hip_bfloat16* W1b     = (__hip_bfloat16*)take((size_t)2 * H_ * H_ * 2);
    __hip_bfloat16* W2b     = (__hip_bfloat16*)take((size_t)H_ * 2 * H_ * 2);
    __hip_bfloat16* Woutb   = (__hip_bfloat16*)take((size_t)32 * H_ * 2);
    __hip_bfloat16* Abuf    = (__hip_bfloat16*)take((size_t)B_ * KC * 2);
    float*          gates   = (float*)         take((size_t)B_ * G4H * 4);
    float*          cbuf    = (float*)         take((size_t)BH * 4);
    float*          ybuf    = (float*)         take((size_t)BH * 4);
    __hip_bfloat16* Asb     = (__hip_bfloat16*)take((size_t)BH * 2);
    __hip_bfloat16* zbuf    = (__hip_bfloat16*)take((size_t)B_ * 2 * H_ * 2);
    float*          kbuf    = (float*)         take((size_t)6 * BH * 4);

    const int TPB = 256;
    auto blks = [](int n, int t) { return (n + t - 1) / t; };

    // ---- one-time weight conversion to bf16 --------------------------------
    convert_wcat_kernel<<<blks(G4H * KC, TPB), TPB, 0, stream>>>(
        W_ih, W_hh, b_ih, b_hh, Wcat, biascat);
    convert_f32_bf16_kernel<<<blks(2 * H_ * H_, TPB), TPB, 0, stream>>>(W1, W1b, 2 * H_ * H_);
    convert_f32_bf16_kernel<<<blks(H_ * 2 * H_, TPB), TPB, 0, stream>>>(W2, W2b, H_ * 2 * H_);
    convert_wout_kernel<<<blks(32 * H_, TPB), TPB, 0, stream>>>(Wout, Woutb);

    // ---- LSTM --------------------------------------------------------------
    lstm_init_kernel<<<blks(B_ * KC, TPB), TPB, 0, stream>>>(x, Abuf, cbuf);
    for (int t = 0; t < L_; ++t) {
        // gates[256,2048] = [x_t|h] @ [W_ih|W_hh]^T + (b_ih+b_hh), K=576
        // 4-wave blocks, block tile 64x128, wave tile 32x64.
        gemm_bf16_kernel<2, 2, 2, 4, 0><<<dim3(G4H / 128, B_ / 64), 128, 0, stream>>>(
            Abuf, KC, Wcat, KC, biascat, gates, nullptr, G4H, KC, G4H);
        lstm_update_kernel<<<blks(BH, TPB), TPB, 0, stream>>>(
            gates, cbuf, Abuf, ybuf, x, t);
    }

    // ---- Neural ODE: 40 fixed dopri5 steps ---------------------------------
    static const float SC[6][5] = {
        {0, 0, 0, 0, 0},
        {1.0f / 5.0f, 0, 0, 0, 0},
        {3.0f / 40.0f, 9.0f / 40.0f, 0, 0, 0},
        {44.0f / 45.0f, -56.0f / 15.0f, 32.0f / 9.0f, 0, 0},
        {19372.0f / 6561.0f, -25360.0f / 2187.0f, 64448.0f / 6561.0f, -212.0f / 729.0f, 0},
        {9017.0f / 3168.0f, -355.0f / 33.0f, 46732.0f / 5247.0f, 49.0f / 176.0f,
         -5103.0f / 18656.0f}};

    for (int step = 0; step < 40; ++step) {
        for (int s = 0; s < 6; ++s) {
            ode_combine_kernel<<<blks(BH, TPB), TPB, 0, stream>>>(
                ybuf, kbuf, Asb, SC[s][0], SC[s][1], SC[s][2], SC[s][3], SC[s][4], s);
            // z[256,1024] = tanh(As @ W1^T + b1)
            gemm_bf16_kernel<2, 2, 2, 4, 1><<<dim3(2 * H_ / 128, B_ / 64), 128, 0, stream>>>(
                Asb, H_, W1b, H_, b1, nullptr, zbuf, 2 * H_, H_, 2 * H_);
            // k_s[256,512] = z @ W2^T + b2
            gemm_bf16_kernel<2, 2, 2, 4, 0><<<dim3(H_ / 128, B_ / 64), 128, 0, stream>>>(
                zbuf, 2 * H_, W2b, 2 * H_, b2, kbuf + (size_t)s * BH, nullptr,
                H_, 2 * H_, H_);
        }
        ode_final_kernel<<<blks(BH, TPB), TPB, 0, stream>>>(ybuf, kbuf);
    }

    // ---- output head: out[256,24] = y @ Wout^T + bout (N padded to 32) -----
    ode_combine_kernel<<<blks(BH, TPB), TPB, 0, stream>>>(
        ybuf, kbuf, Asb, 0.f, 0.f, 0.f, 0.f, 0.f, 0);   // As = bf16(y)
    gemm_bf16_kernel<2, 1, 2, 2, 0><<<dim3(1, B_ / 64), 64, 0, stream>>>(
        Asb, H_, Woutb, H_, bout, (float*)d_out, nullptr, F_, H_, F_);
}
